// GCNLOGSIG_55473797595736
// MI455X (gfx1250) — matrix operand order repaired and checked by reading, not compiled, not run on hardware
//
#include <hip/hip_runtime.h>
#include <math.h>

#define VJ    25
#define TP    304     // T padded to 19*16
#define NMB   8
#define BSEQ  200     // NM*V
#define DIN   4752
#define DINP  4768    // padded K for bf16 WMMA (multiple of 32)
#define G4    384
#define ROWS  10000
#define CHUNK 2000
#define NCHUNK 5
#define MPAD  2016    // chunk rows padded to 63*32
#define HROWS 208     // 200 padded to 13*16

typedef __attribute__((ext_vector_type(16))) __bf16 v16bf;
typedef __attribute__((ext_vector_type(8)))  float  v8f;

union Frag { v16bf v; unsigned int u[8]; };

__device__ __forceinline__ unsigned short f2bf(float f) {
  unsigned int u = __float_as_uint(f);
  u += 0x7FFFu + ((u >> 16) & 1u);           // round-to-nearest-even
  return (unsigned short)(u >> 16);
}

__device__ __forceinline__ v8f wmma_bf16(const Frag& a, const Frag& b, v8f c) {
  return __builtin_amdgcn_wmma_f32_16x16x32_bf16(false, a.v, false, b.v,
                                                 (short)0, c, false, false);
}

// A fragment: 16x32 bf16, row base per lane (M = lane&15), K pairs per ISA layout
__device__ __forceinline__ void load_a(Frag& f, const unsigned short* rowbase,
                                       int kb, int hi) {
#pragma unroll
  for (int i = 0; i < 8; ++i) {
    int kk = kb + ((i >> 2) << 4) + (hi << 3) + ((i & 3) << 1);
    f.u[i] = *(const unsigned int*)(rowbase + kk);
  }
}
// B fragment: 32x16 bf16; colbase = weight row for this lane's N column
__device__ __forceinline__ void load_b(Frag& f, const unsigned short* colbase,
                                       int kb, int hi) {
#pragma unroll
  for (int i = 0; i < 8; ++i)
    f.u[i] = *(const unsigned int*)(colbase + kb + ((i >> 2) << 4) + (hi << 3) + ((i & 3) << 1));
}

// ---------- stage 0: fold A-powers + W_gcn + BN2 into W2 (bf16), fused bias ----------
__global__ void fold_w2_kernel(const float* A_powers, const float* A_res,
                               const float* W_gcn, const float* b_gcn,
                               const float* g2, const float* b2,
                               const float* m2, const float* v2,
                               unsigned short* W2bf, float* bias2) {
  int v = blockIdx.x;                       // joint 0..24
  for (int idx = threadIdx.x; idx < 96 * 96; idx += blockDim.x) {
    int o = idx / 96, k = idx % 96;
    unsigned short val = 0;
    if (k < 75) {
      int c = k / 25, u = k % 25;
      float s2 = g2[o] * rsqrtf(v2[o] + 1e-5f);
      float sum = 0.f;
      for (int kk = 0; kk < 13; ++kk) {
        int ai = (kk * 25 + v) * 25 + u;
        sum += W_gcn[o * 39 + kk * 3 + c] * (A_powers[ai] + A_res[ai]);
      }
      val = f2bf(s2 * sum);
    }
    W2bf[(v * 96 + o) * 96 + k] = val;
  }
  if (v == 0 && threadIdx.x < 96) {
    int o = threadIdx.x;
    float s2 = g2[o] * rsqrtf(v2[o] + 1e-5f);
    bias2[o] = s2 * (b_gcn[o] - m2[o]) + b2[o];
  }
}

// ---------- stage 1: BN1 + repack x to [nm][t(pad)][k=c*25+u] bf16 ----------
__global__ void bnx_kernel(const float* x, const float* bg, const float* bb,
                           const float* bm, const float* bv, unsigned short* xbf) {
  int idx = blockIdx.x * blockDim.x + threadIdx.x;
  int total = NMB * TP * 96;
  if (idx >= total) return;
  int k = idx % 96, t = (idx / 96) % TP, nm = idx / (96 * TP);
  unsigned short out = 0;
  if (t < 300 && k < 75) {
    int c = k / 25, u = k % 25;
    int m = nm % 2;
    int ch = (m * 25 + u) * 3 + c;
    float val = x[((nm * 3 + c) * 300 + t) * 25 + u];
    val = (val - bm[ch]) * (bg[ch] * rsqrtf(bv[ch] + 1e-5f)) + bb[ch];
    out = f2bf(val);
  }
  xbf[idx] = out;
}

// ---------- stage 2: convert LSTM weights to bf16 (K-padded) ----------
__global__ void cvtw_kernel(const float* W_ih, const float* W_hh,
                            unsigned short* Wihbf, unsigned short* Whhbf) {
  int idx = blockIdx.x * blockDim.x + threadIdx.x;
  int tih = G4 * DINP;
  if (idx < tih) {
    int g = idx / DINP, k = idx % DINP;
    Wihbf[idx] = (k < DIN) ? f2bf(W_ih[g * DIN + k]) : (unsigned short)0;
  } else {
    int j = idx - tih;
    if (j < G4 * 96) Whhbf[j] = f2bf(W_hh[j]);
  }
}

// ---------- stage 3: fused GCN GEMM (WMMA) + bias + ReLU -> p[b][t][o] ----------
__global__ void gcn_kernel(const unsigned short* xbf, const unsigned short* W2bf,
                           const float* bias2, float* p) {
  int tt = blockIdx.x;                        // t tile 0..18
  int b  = blockIdx.y;                        // 0..199
  int nm = b / VJ, v = b % VJ;
  int lane = threadIdx.x;
  int m16 = lane & 15, hi = lane >> 4;
  int t0 = tt * 16;
  v8f acc[6] = {};
  const unsigned short* arow = xbf + (size_t)(nm * TP + t0 + m16) * 96;
#pragma unroll
  for (int kc = 0; kc < 3; ++kc) {
    Frag a; load_a(a, arow, kc * 32, hi);
#pragma unroll
    for (int col = 0; col < 6; ++col) {
      Frag bf;
      const unsigned short* wb = W2bf + (size_t)(v * 96 + col * 16 + m16) * 96;
      load_b(bf, wb, kc * 32, hi);
      acc[col] = wmma_bf16(a, bf, acc[col]);
    }
  }
#pragma unroll
  for (int col = 0; col < 6; ++col) {
    int o = col * 16 + m16;
    float bo = bias2[o];
#pragma unroll
    for (int r = 0; r < 8; ++r) {
      int t = t0 + r + 8 * hi;
      if (t < 300) {
        float vl = acc[col][r] + bo;
        p[(size_t)(b * 300 + t) * 96 + o] = vl > 0.f ? vl : 0.f;
      }
    }
  }
}

// ---------- stage 4: log-signature features (start, s1, Levy area) -> bf16 feat ----------
__global__ void feat_kernel(const float* p, unsigned short* featbuf, int chunk) {
  int lr = blockIdx.x;
  int grow = chunk * CHUNK + lr;
  int b = grow / 50, s = grow % 50;
  int start = (int)rint(1.0 + s * (299.0 / 50.0)) - 1;          // RNE == python round
  int nxt   = (int)rint(1.0 + (s + 1) * (299.0 / 50.0)) - 1;
  int len = nxt - start;
  __shared__ float dsh[6][96];
  __shared__ float psh[6][96];
  int c = threadIdx.x;                                          // 0..95
  const float* pb = p + (size_t)b * 300 * 96;
  float run = 0.f;
#pragma unroll
  for (int l = 0; l < 6; ++l) {
    int ii = start + l; if (ii > 298) ii = 298;
    float d = (l < len) ? (pb[(ii + 1) * 96 + c] - pb[ii * 96 + c]) : 0.f;
    psh[l][c] = run; dsh[l][c] = d; run += d;
  }
  unsigned short* fr = featbuf + (size_t)lr * DINP;
  fr[c]      = f2bf(pb[start * 96 + c]);   // start_pos
  fr[96 + c] = f2bf(run);                  // s1
  if (c < 16) fr[DIN + c] = 0;             // K pad
  __syncthreads();
  for (int pi = c; pi < 4560; pi += 96) {
    float fp = (float)pi;
    int cc = (int)((191.0f - sqrtf(191.0f * 191.0f - 8.0f * fp)) * 0.5f);
    if (cc < 0) cc = 0; if (cc > 94) cc = 94;
    while (cc < 94 && (cc + 1) * 95 - ((cc + 1) * cc) / 2 <= pi) ++cc;
    while (cc > 0 && cc * 95 - (cc * (cc - 1)) / 2 > pi) --cc;
    int base = cc * 95 - (cc * (cc - 1)) / 2;
    int dd = cc + 1 + (pi - base);
    float q = 0.f;
#pragma unroll
    for (int l = 0; l < 6; ++l)
      q += psh[l][cc] * dsh[l][dd] - psh[l][dd] * dsh[l][cc];
    fr[192 + pi] = f2bf(0.5f * q);
  }
}

// ---------- stage 5: dominant GEMM feat @ W_ih^T (bf16 WMMA, 32x32 per wave) ----------
__global__ void xw_gemm_kernel(const unsigned short* featbuf, const unsigned short* Wihbf,
                               const float* b_ih, const float* b_hh,
                               float* xw, int chunk) {
  int nt = blockIdx.x;                       // 0..11 (N tiles of 32)
  int mt = blockIdx.y;                       // 0..62 (M tiles of 32)
  int lane = threadIdx.x;
  int m16 = lane & 15, hi = lane >> 4;
  const unsigned short* ar0 = featbuf + (size_t)(mt * 32 + m16) * DINP;
  const unsigned short* ar1 = ar0 + (size_t)16 * DINP;
  const unsigned short* br0 = Wihbf + (size_t)(nt * 32 + m16) * DINP;
  const unsigned short* br1 = br0 + (size_t)16 * DINP;
  v8f a00 = {}, a01 = {}, a10 = {}, a11 = {};
  for (int kc = 0; kc < DINP / 32; ++kc) {
    int kb = kc * 32;
    __builtin_prefetch(ar0 + kb + 256, 0, 1);
    __builtin_prefetch(br0 + kb + 256, 0, 1);
    Frag a0, a1, b0, b1;
    load_a(a0, ar0, kb, hi);
    load_a(a1, ar1, kb, hi);
    load_b(b0, br0, kb, hi);
    load_b(b1, br1, kb, hi);
    a00 = wmma_bf16(a0, b0, a00);
    a01 = wmma_bf16(a0, b1, a01);
    a10 = wmma_bf16(a1, b0, a10);
    a11 = wmma_bf16(a1, b1, a11);
  }
  int g0 = nt * 32 + m16, g1 = g0 + 16;
  float bias0 = b_ih[g0] + b_hh[g0];
  float bias1 = b_ih[g1] + b_hh[g1];
#pragma unroll
  for (int r = 0; r < 8; ++r) {
    int row0 = mt * 32 + r + 8 * hi;
    int row1 = row0 + 16;
    if (row0 < CHUNK) {
      size_t o = (size_t)(chunk * CHUNK + row0) * G4;
      xw[o + g0] = a00[r] + bias0;
      xw[o + g1] = a01[r] + bias1;
    }
    if (row1 < CHUNK) {
      size_t o = (size_t)(chunk * CHUNK + row1) * G4;
      xw[o + g0] = a10[r] + bias0;
      xw[o + g1] = a11[r] + bias1;
    }
  }
}

// ---------- stage 6: LSTM ----------
__global__ void lstm_init_kernel(unsigned short* hbf, float* cbuf) {
  int idx = blockIdx.x * blockDim.x + threadIdx.x;
  if (idx < HROWS * 96) hbf[idx] = 0;
  if (idx < BSEQ * 96) cbuf[idx] = 0.f;
}

__global__ void lstm_gemm_kernel(const unsigned short* hbf, const unsigned short* Whhbf,
                                 const float* xw, float* gbuf, int s) {
  int nt = blockIdx.x;                       // 0..23
  int mt = blockIdx.y;                       // 0..12
  int lane = threadIdx.x;
  int m16 = lane & 15, hi = lane >> 4;
  const unsigned short* arow = hbf + (size_t)(mt * 16 + m16) * 96;
  const unsigned short* brow = Whhbf + (size_t)(nt * 16 + m16) * 96;
  v8f acc = {};
#pragma unroll
  for (int kc = 0; kc < 3; ++kc) {
    Frag a, b;
    load_a(a, arow, kc * 32, hi);
    load_b(b, brow, kc * 32, hi);
    acc = wmma_bf16(a, b, acc);
  }
  int g = nt * 16 + m16;
#pragma unroll
  for (int r = 0; r < 8; ++r) {
    int row = mt * 16 + r + 8 * hi;
    if (row < BSEQ)
      gbuf[(size_t)row * G4 + g] = acc[r] + xw[((size_t)row * 50 + s) * G4 + g];
  }
}

__device__ __forceinline__ float sigf(float x) { return 1.f / (1.f + __expf(-x)); }

__global__ void lstm_update_kernel(const float* gbuf, float* cbuf,
                                   unsigned short* hbf, float* hs, int s) {
  int b = blockIdx.x; int j = threadIdx.x;   // 200 x 96
  const float* g = gbuf + (size_t)b * G4;
  float i_ = g[j], f_ = g[96 + j], gg = g[192 + j], o_ = g[288 + j];
  float c = cbuf[b * 96 + j];
  c = sigf(f_) * c + sigf(i_) * tanhf(gg);
  float h = sigf(o_) * tanhf(c);
  cbuf[b * 96 + j] = c;
  hbf[b * 96 + j] = f2bf(h);
  hs[((size_t)b * 50 + s) * 96 + j] = h;
}

// ---------- stage 7: mean over (m, v, seg) + FC ----------
__global__ void reduce_fc_kernel(const float* hs, const float* W_fc,
                                 const float* b_fc, float* out) {
  int n = blockIdx.x; int c = threadIdx.x;   // 4 x 96
  __shared__ float avg[96];
  float acc = 0.f;
  for (int m = 0; m < 2; ++m)
    for (int v = 0; v < VJ; ++v) {
      int b = (n * 2 + m) * VJ + v;
      const float* hb = hs + (size_t)b * 50 * 96;
      for (int s = 0; s < 50; ++s) acc += hb[s * 96 + c];
    }
  avg[c] = acc * (1.f / 2500.f);
  __syncthreads();
  if (c < 60) {
    float o = b_fc[c];
    for (int k = 0; k < 96; ++k) o += avg[k] * W_fc[c * 96 + k];
    out[n * 60 + c] = o;
  }
}

extern "C" void kernel_launch(void* const* d_in, const int* in_sizes, int n_in,
                              void* d_out, int out_size, void* d_ws, size_t ws_size,
                              hipStream_t stream) {
  (void)in_sizes; (void)n_in; (void)out_size; (void)ws_size;
  const float* x    = (const float*)d_in[0];
  const float* bn_g = (const float*)d_in[1];
  const float* bn_b = (const float*)d_in[2];
  const float* bn_m = (const float*)d_in[3];
  const float* bn_v = (const float*)d_in[4];
  const float* A_pw = (const float*)d_in[5];
  const float* A_rs = (const float*)d_in[6];
  const float* W_gc = (const float*)d_in[7];
  const float* b_gc = (const float*)d_in[8];
  const float* g2   = (const float*)d_in[9];
  const float* b2   = (const float*)d_in[10];
  const float* m2   = (const float*)d_in[11];
  const float* v2   = (const float*)d_in[12];
  const float* W_ih = (const float*)d_in[13];
  const float* W_hh = (const float*)d_in[14];
  const float* b_ih = (const float*)d_in[15];
  const float* b_hh = (const float*)d_in[16];
  const float* W_fc = (const float*)d_in[17];
  const float* b_fc = (const float*)d_in[18];

  char* ws = (char*)d_ws;
  size_t off = 0;
  auto carve = [&](size_t bytes) -> char* {
    char* q = ws + off;
    off = (off + bytes + 255) & ~(size_t)255;
    return q;
  };
  unsigned short* W2bf  = (unsigned short*)carve((size_t)VJ * 96 * 96 * 2);
  float*          bias2 = (float*)carve(96 * 4);
  unsigned short* xbf   = (unsigned short*)carve((size_t)NMB * TP * 96 * 2);
  float*          p     = (float*)carve((size_t)BSEQ * 300 * 96 * 4);
  unsigned short* Wihbf = (unsigned short*)carve((size_t)G4 * DINP * 2);
  unsigned short* Whhbf = (unsigned short*)carve((size_t)G4 * 96 * 2);
  unsigned short* featb = (unsigned short*)carve((size_t)MPAD * DINP * 2);
  float*          xw    = (float*)carve((size_t)ROWS * G4 * 4);
  float*          gbuf  = (float*)carve((size_t)HROWS * G4 * 4);
  unsigned short* hbf   = (unsigned short*)carve((size_t)HROWS * 96 * 2);
  float*          cbuf  = (float*)carve((size_t)BSEQ * 96 * 4);
  float*          hs    = (float*)carve((size_t)BSEQ * 50 * 96 * 4);

  fold_w2_kernel<<<VJ, 256, 0, stream>>>(A_pw, A_rs, W_gc, b_gc, g2, b2, m2, v2,
                                         W2bf, bias2);
  {
    int total = NMB * TP * 96;
    bnx_kernel<<<(total + 255) / 256, 256, 0, stream>>>(x, bn_g, bn_b, bn_m, bn_v, xbf);
  }
  {
    int total = G4 * DINP + G4 * 96;
    cvtw_kernel<<<(total + 255) / 256, 256, 0, stream>>>(W_ih, W_hh, Wihbf, Whhbf);
  }
  gcn_kernel<<<dim3(19, BSEQ), 32, 0, stream>>>(xbf, W2bf, bias2, p);

  for (int chunk = 0; chunk < NCHUNK; ++chunk) {
    feat_kernel<<<CHUNK, 96, 0, stream>>>(p, featb, chunk);
    xw_gemm_kernel<<<dim3(G4 / 32, MPAD / 32), 32, 0, stream>>>(featb, Wihbf,
                                                                b_ih, b_hh, xw, chunk);
  }

  lstm_init_kernel<<<(HROWS * 96 + 255) / 256, 256, 0, stream>>>(hbf, cbuf);
  for (int s = 0; s < 50; ++s) {
    lstm_gemm_kernel<<<dim3(G4 / 16, HROWS / 16), 32, 0, stream>>>(hbf, Whhbf, xw, gbuf, s);
    lstm_update_kernel<<<BSEQ, 96, 0, stream>>>(gbuf, cbuf, hbf, hs, s);
  }

  reduce_fc_kernel<<<4, 96, 0, stream>>>(hs, W_fc, b_fc, (float*)d_out);
}
